// SequencePredictorRNN_33174327394420
// MI455X (gfx1250) — compile-verified, matrix operand order checked by reading, not run on hardware
//
#include <hip/hip_runtime.h>

#define D_MODEL   1024
#define N_CLASSES 512
#define SEQL      1024
#define BATCH     128
#define G4        (4 * D_MODEL)
#define NWG_REC   32

typedef __attribute__((ext_vector_type(16))) __bf16 v16bf;
typedef __attribute__((ext_vector_type(8)))  float  v8f;
typedef __attribute__((ext_vector_type(4)))  float  f32x4;
typedef __attribute__((ext_vector_type(4)))  __bf16 bf16x4;

union FragU {
    v16bf  f;
    f32x4  q[2];
    __bf16 e[16];
};

// ---------------------------------------------------------------------------
// Fragment loaders. CDNA5 16-bit A layout (16x32): lane L holds row M=L%16;
// lanes 0-15 hold K = {0..7, 16..23}, lanes 16-31 hold K = {8..15, 24..31}.
// B (32x16) mirrors this with N in place of M, so both A (row-major MxK) and
// B (row-major NxK, i.e. W for gates = h @ W^T) load identically.
// ---------------------------------------------------------------------------
__device__ inline v16bf load_frag_bf16(const __bf16* base, int ld) {
    const int lane = threadIdx.x & 31;
    const __bf16* p = base + (size_t)(lane & 15) * (size_t)ld + ((lane >> 4) << 3);
    FragU u;
    u.q[0] = *(const f32x4*)(p);
    u.q[1] = *(const f32x4*)(p + 16);
    return u.f;
}

// A fragment gathered from f32 rows (embedding lookup), converted to bf16.
// p already includes this lane's row pointer and half-offset.
__device__ inline v16bf cvt_frag_f32(const float* p) {
    FragU u;
#pragma unroll
    for (int i = 0; i < 8; ++i) u.e[i] = (__bf16)p[i];
#pragma unroll
    for (int i = 0; i < 8; ++i) u.e[8 + i] = (__bf16)p[16 + i];
    return u.f;
}

// ---------------------------------------------------------------------------
// Async global -> LDS copy (gfx1250). 16 bytes per lane per call.
// ---------------------------------------------------------------------------
__device__ inline void cp_b128_g2l(void* l, const void* g) {
    unsigned loff = (unsigned)(size_t)(const __attribute__((address_space(3))) void*)l;
    unsigned long long ga = (unsigned long long)g;
    asm volatile("global_load_async_to_lds_b128 %0, %1, off"
                 :: "v"(loff), "v"(ga) : "memory");
}
__device__ inline void cp_wait_async() {
    asm volatile("s_wait_asynccnt 0" ::: "memory");
}

// ---------------------------------------------------------------------------
// Shared building block: one d-slice (wg) x one m-block (wave) input-projection
// GEMM: acc[j] += bf16(emb[x[t,m]]) @ Wih^T tile j, j = 2*q + hh.
// ---------------------------------------------------------------------------
__device__ inline void xpart_gemm(v8f acc[8], const int* __restrict__ x,
                                  const float* __restrict__ emb,
                                  const __bf16* __restrict__ Wih,
                                  int wg, int t, int mbase, int lane15, int halfoff)
{
    const int xi = x[t * BATCH + mbase + lane15];
    const float* arow = emb + (size_t)xi * D_MODEL + halfoff;
    const __bf16* wB[8];
#pragma unroll
    for (int j = 0; j < 8; ++j)
        wB[j] = Wih + (size_t)((j >> 1) * D_MODEL + wg * 32 + (j & 1) * 16) * D_MODEL;
#pragma unroll 2
    for (int kc = 0; kc < D_MODEL / 32; ++kc) {
        v16bf af = cvt_frag_f32(arow + kc * 32);
#pragma unroll
        for (int j = 0; j < 8; ++j) {
            v16bf bf = load_frag_bf16(wB[j] + kc * 32, D_MODEL);
            acc[j] = __builtin_amdgcn_wmma_f32_16x16x32_bf16(
                false, af, false, bf, (short)0, acc[j], false, false);
        }
    }
}

// xg is stored in WMMA C-tile lane layout: float index =
//   ((((t*32 + wg)*8 + w)*8 + j)*32 + lane)*8 + v
__device__ inline size_t xg_base_idx(int t, int wg, int w, int lane) {
    return ((((size_t)t * 32 + wg) * 8 + w) * 8) * 256 + (size_t)lane * 8;
}

__device__ inline void load_xg_tiles(v8f acc[8], const float* __restrict__ xg,
                                     int t, int wg, int w, int lane) {
    const float* base = xg + xg_base_idx(t, wg, w, lane);
#pragma unroll
    for (int j = 0; j < 8; ++j)
        acc[j] = *(const v8f*)(base + (size_t)j * 256);
}

// ---------------------------------------------------------------------------
// Small prep kernels
// ---------------------------------------------------------------------------
__global__ void cast_f32_to_bf16_kernel(const float* __restrict__ src,
                                        __bf16* __restrict__ dst, int n4) {
    int i = blockIdx.x * blockDim.x + threadIdx.x;
    if (i < n4) {
        f32x4 v = ((const f32x4*)src)[i];
        bf16x4 o;
#pragma unroll
        for (int e = 0; e < 4; ++e) o[e] = (__bf16)v[e];
        ((bf16x4*)dst)[i] = o;
    }
}

__global__ void bias_sum_kernel(const float* __restrict__ a,
                                const float* __restrict__ b,
                                float* __restrict__ o, int n) {
    int i = blockIdx.x * blockDim.x + threadIdx.x;
    if (i < n) o[i] = a[i] + b[i];
}

// ---------------------------------------------------------------------------
// Fully parallel input-projection precompute: xg[t] = emb[x[t]] @ Wih^T
// (stored f32 in WMMA C-tile lane layout). Grid (SEQL, 32), 256 threads.
// Removes half the work from the serialized scan's critical path.
// ---------------------------------------------------------------------------
__global__ __launch_bounds__(256, 2) void xg_precompute_kernel(
    const int*    __restrict__ x,
    const float*  __restrict__ emb,
    const __bf16* __restrict__ Wih,
    float*        __restrict__ xg)
{
    const int t       = blockIdx.x;
    const int wg      = blockIdx.y;
    const int w       = threadIdx.x >> 5;
    const int lane    = threadIdx.x & 31;
    const int lane15  = lane & 15;
    const int halfoff = (lane >> 4) << 3;
    const int mbase   = w * 16;

    v8f acc[8] = {};
    xpart_gemm(acc, x, emb, Wih, wg, t, mbase, lane15, halfoff);

    float* base = xg + xg_base_idx(t, wg, w, lane);
#pragma unroll
    for (int j = 0; j < 8; ++j)
        *(v8f*)(base + (size_t)j * 256) = acc[j];
}

// ---------------------------------------------------------------------------
// Persistent LSTM scan. Grid = 32 workgroups x 256 threads (8 waves).
// Workgroup wg owns d-slice [32*wg, 32*wg+32) -> 8 N-tiles (4 gates x 2).
// Wave w owns M rows [16w, 16w+16). c state lives in registers.
// The workgroup's entire W_hh slice (8 tiles x 1024 K = 256 KB) is staged in
// LDS once via async copies -- it fits the WGP's 320 KB LDS -- so all 1024
// steps read B operands with ds_load_b128 instead of re-reading L2.
// Pipeline per step: wait(t) -> h-part GEMM (A prefetched one k-chunk ahead)
// -> gates/epilogue -> signal -> input part for t+1 (hides barrier latency).
// USE_XG: input part is 16 b128 loads of precomputed tiles; else fused GEMM.
// ---------------------------------------------------------------------------
template <bool USE_XG>
__global__ __launch_bounds__(256, 1) void lstm_scan_kernel(
    const int*    __restrict__ x,
    const float*  __restrict__ emb,
    const __bf16* __restrict__ Wih,
    const __bf16* __restrict__ Whh,
    const float*  __restrict__ bc,
    const float*  __restrict__ xg,
    __bf16*       __restrict__ ys,
    unsigned*                  bar)
{
    __shared__ __bf16 WhhS[8][32][16 * 32];   // 256 KB: full W_hh slice of this WG

    const int wg      = blockIdx.x;            // d-slice
    const int w       = threadIdx.x >> 5;      // wave -> m block
    const int lane    = threadIdx.x & 31;
    const int lane15  = lane & 15;
    const int halfoff = (lane >> 4) << 3;      // 0 or 8
    const int mbase   = w * 16;

    // ---- stage W_hh slice into LDS (async), chunked as [j][kc][row][32k] ----
    for (int u = threadIdx.x; u < 16384; u += 256) {
        const int seg = u & 3;
        const int r   = (u >> 2) & 15;
        const int kc  = (u >> 6) & 31;
        const int j   = u >> 11;
        const __bf16* g = Whh
            + (size_t)((j >> 1) * D_MODEL + wg * 32 + (j & 1) * 16 + r) * D_MODEL
            + kc * 32 + seg * 8;
        cp_b128_g2l(&WhhS[j][kc][r * 32 + seg * 8], g);
    }

    float bias[4][2];
#pragma unroll
    for (int q = 0; q < 4; ++q)
#pragma unroll
        for (int hh = 0; hh < 2; ++hh)
            bias[q][hh] = bc[q * D_MODEL + wg * 32 + hh * 16 + lane15];

    float creg[8][2];
#pragma unroll
    for (int v = 0; v < 8; ++v) { creg[v][0] = 0.f; creg[v][1] = 0.f; }

    const v8f zero8 = {};

    // pending input-part accumulator for step 0 (overlaps the LDS staging)
    v8f accP[8];
#pragma unroll
    for (int j = 0; j < 8; ++j) accP[j] = zero8;
    if constexpr (USE_XG) load_xg_tiles(accP, xg, 0, wg, w, lane);
    else xpart_gemm(accP, x, emb, Wih, wg, 0, mbase, lane15, halfoff);

    cp_wait_async();
    __syncthreads();   // W_hh slice resident in LDS from here on

    for (int t = 0; t < SEQL; ++t) {
        // ---- wait for step t-1 to be globally complete, then h-part ----
        if (t > 0) {
            if (threadIdx.x == 0) {
                const unsigned tgt = (unsigned)t * NWG_REC;
                while (__hip_atomic_load(bar, __ATOMIC_ACQUIRE,
                                         __HIP_MEMORY_SCOPE_AGENT) < tgt)
                    __builtin_amdgcn_s_sleep(1);
            }
            __syncthreads();
            __threadfence();   // acquire: make ys[t-1] visible to this WGP

            const __bf16* hbase = ys + ((size_t)(t - 1) * BATCH + mbase) * D_MODEL;
            v16bf af = load_frag_bf16(hbase, D_MODEL);
            for (int kc = 0; kc < D_MODEL / 32; ++kc) {
                v16bf afn;
                if (kc + 1 < D_MODEL / 32)
                    afn = load_frag_bf16(hbase + (kc + 1) * 32, D_MODEL);
#pragma unroll
                for (int j = 0; j < 8; ++j) {
                    v16bf bf = load_frag_bf16(WhhS[j][kc], 32);   // ds_load_b128
                    accP[j] = __builtin_amdgcn_wmma_f32_16x16x32_bf16(
                        false, af, false, bf, (short)0, accP[j], false, false);
                }
                af = afn;
            }
        }

        // ---- gates + cell update; write h_t as bf16 ----
        // C/D layout: VGPR v -> M = v + 8*(lane>=16); N = lane%16.
#pragma unroll
        for (int v = 0; v < 8; ++v) {
            const int mrow = mbase + v + halfoff;
            __bf16* yrow = ys + ((size_t)t * BATCH + mrow) * D_MODEL;
#pragma unroll
            for (int hh = 0; hh < 2; ++hh) {
                float ir  = accP[0 + hh][v] + bias[0][hh];
                float fr  = accP[2 + hh][v] + bias[1][hh];
                float gr  = accP[4 + hh][v] + bias[2][hh];
                float orr = accP[6 + hh][v] + bias[3][hh];
                float ig = 1.f / (1.f + __expf(-ir));
                float fg = 1.f / (1.f + __expf(-fr));
                float gg = tanhf(gr);
                float og = 1.f / (1.f + __expf(-orr));
                float c = fg * creg[v][hh] + ig * gg;
                creg[v][hh] = c;
                float h = og * tanhf(c);
                yrow[wg * 32 + hh * 16 + lane15] = (__bf16)h;
            }
        }

        // ---- publish step t, then overlap next step's input part ----
        __threadfence();
        __syncthreads();
        if (threadIdx.x == 0)
            __hip_atomic_fetch_add(bar, 1u, __ATOMIC_ACQ_REL, __HIP_MEMORY_SCOPE_AGENT);

        if (t + 1 < SEQL) {
#pragma unroll
            for (int j = 0; j < 8; ++j) accP[j] = zero8;
            if constexpr (USE_XG) load_xg_tiles(accP, xg, t + 1, wg, w, lane);
            else xpart_gemm(accP, x, emb, Wih, wg, t + 1, mbase, lane15, halfoff);
        }
    }
}

// ---------------------------------------------------------------------------
// logits[b,s,n] = sum_d ys[s,b,d] * Wp[n,d] + bp[n]
// Block: 128 rows (8 waves x 16) x 64 cols; W_pred tiles double-buffered in
// LDS via async global->LDS copies; fragments read back with ds_load_b128.
// ---------------------------------------------------------------------------
__global__ __launch_bounds__(256, 2) void logits_kernel(
    const __bf16* __restrict__ ys,
    const __bf16* __restrict__ Wp,
    const float*  __restrict__ bp,
    float*        __restrict__ out)
{
    __shared__ __bf16 Bs[2][64 * 32];   // 8 KB
    const int lane    = threadIdx.x & 31;
    const int lane15  = lane & 15;
    const int halfoff = (lane >> 4) << 3;
    const int w       = threadIdx.x >> 5;
    const int rbase   = blockIdx.x * 128 + w * 16;   // row in (s*128+b)
    const int nbase   = blockIdx.y * 64;

    v8f acc[4] = {};

    const int crow = threadIdx.x >> 2;        // 0..63 : n row within tile
    const int cseg = (threadIdx.x & 3) * 8;   // 16B segment within 32-elem k chunk
    const __bf16* gsrc = Wp + (size_t)(nbase + crow) * D_MODEL + cseg;
    __bf16* l0 = &Bs[0][crow * 32 + cseg];
    __bf16* l1 = &Bs[1][crow * 32 + cseg];

    cp_b128_g2l(l0, gsrc);
    cp_wait_async();
    __syncthreads();

    float bpv[4];
#pragma unroll
    for (int jt = 0; jt < 4; ++jt) bpv[jt] = bp[nbase + jt * 16 + lane15];

    const __bf16* abase = ys + (size_t)rbase * D_MODEL;

    for (int kc = 0; kc < D_MODEL / 32; ++kc) {
        const int cur = kc & 1;
        if (kc + 1 < D_MODEL / 32)
            cp_b128_g2l(cur ? l0 : l1, gsrc + (size_t)(kc + 1) * 32);

        v16bf af = load_frag_bf16(abase + kc * 32, D_MODEL);
#pragma unroll
        for (int jt = 0; jt < 4; ++jt) {
            v16bf bf = load_frag_bf16(&Bs[cur][jt * 16 * 32], 32);
            acc[jt] = __builtin_amdgcn_wmma_f32_16x16x32_bf16(
                false, af, false, bf, (short)0, acc[jt], false, false);
        }
        cp_wait_async();
        __syncthreads();
    }

#pragma unroll
    for (int v = 0; v < 8; ++v) {
        const int r = rbase + v + halfoff;   // M offset for upper lanes is +8
        const int s = r >> 7;
        const int b = r & 127;
        float* orow = out + ((size_t)b * SEQL + s) * (size_t)N_CLASSES + nbase;
#pragma unroll
        for (int jt = 0; jt < 4; ++jt)
            orow[jt * 16 + lane15] = acc[jt][v] + bpv[jt];
    }
}

// ---------------------------------------------------------------------------
extern "C" void kernel_launch(void* const* d_in, const int* in_sizes, int n_in,
                              void* d_out, int out_size, void* d_ws, size_t ws_size,
                              hipStream_t stream) {
    (void)in_sizes; (void)n_in; (void)out_size;

    const int*   x    = (const int*)d_in[0];
    const float* emb  = (const float*)d_in[1];
    const float* Wih  = (const float*)d_in[2];
    const float* Whh  = (const float*)d_in[3];
    const float* bih  = (const float*)d_in[4];
    const float* bhh  = (const float*)d_in[5];
    const float* Wp   = (const float*)d_in[6];
    const float* bp   = (const float*)d_in[7];
    float*       out  = (float*)d_out;

    unsigned char* w8 = (unsigned char*)d_ws;
    unsigned* bar     = (unsigned*)(w8 + 0);                       // 1 KB
    float*    bc      = (float*)(w8 + 1024);                       // 16 KB
    __bf16*   Wih_bf  = (__bf16*)(w8 + 1024 + 16384);              // 8 MB
    __bf16*   Whh_bf  = Wih_bf + (size_t)G4 * D_MODEL;             // 8 MB
    __bf16*   Wp_bf   = Whh_bf + (size_t)G4 * D_MODEL;             // 1 MB
    __bf16*   ysbf    = Wp_bf + (size_t)N_CLASSES * D_MODEL;       // 256 MB
    float*    xg      = (float*)(ysbf + (size_t)SEQL * BATCH * D_MODEL); // 2 GB

    const size_t base_bytes = 1024 + 16384
        + 2ull * G4 * D_MODEL * 2 + (size_t)N_CLASSES * D_MODEL * 2
        + (size_t)SEQL * BATCH * D_MODEL * 2;
    const size_t xg_bytes = (size_t)SEQL * BATCH * G4 * 4;
    const bool use_xg = ws_size >= base_bytes + xg_bytes;

    hipMemsetAsync(bar, 0, 1024, stream);

    // weight casts (vectorized x4)
    {
        int n4 = (G4 * D_MODEL) / 4;
        cast_f32_to_bf16_kernel<<<(n4 + 255) / 256, 256, 0, stream>>>(Wih, Wih_bf, n4);
        cast_f32_to_bf16_kernel<<<(n4 + 255) / 256, 256, 0, stream>>>(Whh, Whh_bf, n4);
        int n4p = (N_CLASSES * D_MODEL) / 4;
        cast_f32_to_bf16_kernel<<<(n4p + 255) / 256, 256, 0, stream>>>(Wp, Wp_bf, n4p);
        bias_sum_kernel<<<(G4 + 255) / 256, 256, 0, stream>>>(bih, bhh, bc, G4);
    }

    if (use_xg) {
        xg_precompute_kernel<<<dim3(SEQL, 32), 256, 0, stream>>>(x, emb, Wih_bf, xg);
        lstm_scan_kernel<true><<<NWG_REC, 256, 0, stream>>>(
            x, emb, Wih_bf, Whh_bf, bc, xg, ysbf, bar);
    } else {
        lstm_scan_kernel<false><<<NWG_REC, 256, 0, stream>>>(
            x, emb, Wih_bf, Whh_bf, bc, xg, ysbf, bar);
    }

    logits_kernel<<<dim3((BATCH * SEQL) / 128, N_CLASSES / 64), 256, 0, stream>>>(
        ysbf, Wp_bf, bp, out);
}
